// LayerRGAT_3487513444607
// MI455X (gfx1250) — matrix operand order repaired and checked by reading, not compiled, not run on hardware
//
#include <hip/hip_runtime.h>
#include <math.h>

// ---------------- CDNA5 WMMA types & helpers (wave32, gfx1250) ----------------
typedef __attribute__((ext_vector_type(16))) __bf16 v16bf;
typedef __attribute__((ext_vector_type(8)))  __bf16 v8bf;
typedef __attribute__((ext_vector_type(4)))  __bf16 v4bf;
typedef __attribute__((ext_vector_type(8)))  float  v8f;

#define RSA 40   // LDS row stride (elements) for 32-wide K tiles (80B, 16B aligned, bank-spread)
#define RSH 72   // LDS row stride for 64-wide he tile
#define RSB 136  // LDS row stride for 128-wide hs tile

// pack float4 -> 4 bf16 (native converts), single 8-byte LDS store
__device__ __forceinline__ void store4bf(__bf16* dst, float4 v) {
    v4bf p = { (__bf16)v.x, (__bf16)v.y, (__bf16)v.z, (__bf16)v.w };
    *(v4bf*)dst = p;
}

// A fragment (16x32, MxK): lane l -> row m=l&15, k-base b=(l&16)?8:0,
// elements j=0..7 -> k=b+j ; j=8..15 -> k=b+16+(j-8)
__device__ __forceinline__ v16bf load_a_frag(const __bf16* row, int lane) {
    int kb = (lane & 16) ? 8 : 0;
    v8bf lo = *(const v8bf*)(row + kb);
    v8bf hi = *(const v8bf*)(row + kb + 16);
    return __builtin_shufflevector(lo, hi, 0,1,2,3,4,5,6,7,8,9,10,11,12,13,14,15);
}
// B fragment (32x16, KxN) read from LDS stored TRANSPOSED as Bt[n][k]:
// lane l -> col n=l&15, k-base b=(l&16)?8:0, elements j=0..7 -> k=b+j ; j=8..15 -> k=b+16+(j-8)
__device__ __forceinline__ v16bf load_b_frag(const __bf16* rowT, int lane) {
    int kb = (lane & 16) ? 8 : 0;
    v8bf lo = *(const v8bf*)(rowT + kb);
    v8bf hi = *(const v8bf*)(rowT + kb + 16);
    return __builtin_shufflevector(lo, hi, 0,1,2,3,4,5,6,7,8,9,10,11,12,13,14,15);
}

__device__ __forceinline__ void atomicMaxF(float* addr, float val) {
    if (val >= 0.0f) atomicMax((int*)addr, __float_as_int(val));
    else             atomicMin((unsigned int*)addr, __float_as_uint(val));
}
__device__ __forceinline__ float sigmoidf(float x) { return 1.0f / (1.0f + expf(-x)); }

// ---------------- init / small kernels ----------------
__global__ void init_kernel(float* hcat, float* deg, float* maxA, float* sumA,
                            float* maxG, float* sumG, int N_) {
    long i = (long)blockIdx.x * blockDim.x + threadIdx.x;
    long tot = (long)N_ * 256;
    if (i < tot) hcat[i] = 0.0f;
    if (i < N_) { deg[i] = 0.0f; sumA[i] = 0.0f; maxA[i] = -3.402823466e38f; }
    if (i < (long)N_ * 3) { sumG[i] = 0.0f; maxG[i] = -3.402823466e38f; }
}

__global__ void wsum_kernel(const float* W, float* Wsum) {
    int i = blockIdx.x * blockDim.x + threadIdx.x;
    if (i < 128 * 128) Wsum[i] = (W[i] + W[16384 + i] + W[32768 + i]) * (1.0f / 3.0f);
}

// ---------------- generic WMMA GEMM: C = act(A[M,K] @ W[K,N] + bias) ----------------
__global__ __launch_bounds__(256) void gemm_bias_act(const float* __restrict__ A,
                                                     const float* __restrict__ W,
                                                     const float* __restrict__ bias,
                                                     float* __restrict__ C,
                                                     int M_, int K_, int N_, int act) {
    __shared__ __align__(16) __bf16 As[64 * RSA];
    __shared__ __align__(16) __bf16 Bt[32 * RSA];
    int tid = threadIdx.x, lane = tid & 31, wid = tid >> 5;
    int wm = wid >> 1, wn = wid & 1;
    int bm0 = blockIdx.x * 64, bn0 = blockIdx.y * 32;
    bool full = (bm0 + 64 <= M_);
    v8f acc;
#pragma unroll
    for (int i = 0; i < 8; i++) acc[i] = 0.0f;

    for (int k0 = 0; k0 < K_; k0 += 32) {
        // A tile: 64x32 f32 -> bf16 LDS, float4 loads + b64 LDS stores
        if (full) {
#pragma unroll
            for (int i = 0; i < 2; i++) {
                int f = tid + i * 256;               // 512 float4
                int r = f >> 3, c = (f & 7) << 2;
                float4 v = *(const float4*)(A + (size_t)(bm0 + r) * K_ + k0 + c);
                store4bf(&As[r * RSA + c], v);
            }
        } else {
#pragma unroll
            for (int i = 0; i < 2; i++) {
                int f = tid + i * 256;
                int r = f >> 3, c = (f & 7) << 2;
                int gr = bm0 + r;
                float4 v = make_float4(0.f, 0.f, 0.f, 0.f);
                if (gr < M_) v = *(const float4*)(A + (size_t)gr * K_ + k0 + c);
                store4bf(&As[r * RSA + c], v);
            }
        }
        // B tile: 32x32 f32 -> bf16 LDS transposed (Bt[n][k])
        {
            int r = tid >> 3, c = (tid & 7) << 2;    // r=k, c=n  (256 float4)
            float4 v = *(const float4*)(W + (size_t)(k0 + r) * N_ + bn0 + c);
            Bt[(c + 0) * RSA + r] = (__bf16)v.x;
            Bt[(c + 1) * RSA + r] = (__bf16)v.y;
            Bt[(c + 2) * RSA + r] = (__bf16)v.z;
            Bt[(c + 3) * RSA + r] = (__bf16)v.w;
        }
        // prefetch next K tile of A (global_prefetch_b8)
        if (k0 + 32 < K_) {
            int r = tid >> 3, c = (tid & 7) << 2;
            int gr = bm0 + r;
            if (gr < M_) __builtin_prefetch(A + (size_t)gr * K_ + k0 + 32 + c);
        }
        __syncthreads();
        v16bf af = load_a_frag(&As[(wm * 16 + (lane & 15)) * RSA], lane);
        v16bf bf = load_b_frag(&Bt[(wn * 16 + (lane & 15)) * RSA], lane);
        acc = __builtin_amdgcn_wmma_f32_16x16x32_bf16(false, af, false, bf,
                                                      (short)0, acc, false, false);
        __syncthreads();
    }
    int n = bn0 + wn * 16 + (lane & 15);
    float bv = bias ? bias[n] : 0.0f;
    int rbase = bm0 + wm * 16 + ((lane >> 4) << 3);
    if (full) {
#pragma unroll
        for (int i = 0; i < 8; i++) {
            float v = acc[i] + bv;
            if (act) v = v > 0.0f ? v : 0.0f;
            C[(size_t)(rbase + i) * N_ + n] = v;
        }
    } else {
#pragma unroll
        for (int i = 0; i < 8; i++) {
            int m = rbase + i;
            if (m < M_) {
                float v = acc[i] + bv;
                if (act) v = v > 0.0f ? v : 0.0f;
                C[(size_t)m * N_ + n] = v;
            }
        }
    }
}

// ---------------- fused edge gate: g[e,m] = sigmoid(relu(he@W1_m+b1)·W2_m + b2_m) ----------------
__global__ __launch_bounds__(256) void edge_gate_kernel(const float* __restrict__ he,
                                                        const float* __restrict__ W1,  // [3,64,128]
                                                        const float* __restrict__ b1,  // [3,128]
                                                        const float* __restrict__ W2,  // [3,128]
                                                        const float* __restrict__ b2,  // [3]
                                                        float* __restrict__ g, int E_) {
    __shared__ __align__(16) __bf16 Hes[64 * RSH];
    __shared__ __align__(16) __bf16 Bt[32 * RSA];
    __shared__ float gpart[64];
    int tid = threadIdx.x, lane = tid & 31, wid = tid >> 5;
    int wm = wid >> 1, wn = wid & 1;
    int e0 = blockIdx.x * 64;

    // he tile 64x64: float4 loads + b64 LDS stores
#pragma unroll
    for (int i = 0; i < 4; i++) {
        int f = tid + i * 256;                   // 1024 float4
        int r = f >> 4, c = (f & 15) << 2;
        int ge = e0 + r;
        float4 v = make_float4(0.f, 0.f, 0.f, 0.f);
        if (ge < E_) v = *(const float4*)(he + (size_t)ge * 64 + c);
        store4bf(&Hes[r * RSH + c], v);
    }
    __syncthreads();

    for (int m = 0; m < 3; m++) {
        if (tid < 64) gpart[tid] = 0.0f;
        __syncthreads();
        for (int nt = 0; nt < 4; nt++) {
            int n0 = nt * 32;
            v8f acc;
#pragma unroll
            for (int i = 0; i < 8; i++) acc[i] = 0.0f;
            for (int kt = 0; kt < 2; kt++) {
                int k0 = kt * 32;
                {
                    int r = tid >> 3, c = (tid & 7) << 2;  // r=k, c=n
                    float4 v = *(const float4*)(W1 + ((size_t)m * 64 + k0 + r) * 128 + n0 + c);
                    Bt[(c + 0) * RSA + r] = (__bf16)v.x;
                    Bt[(c + 1) * RSA + r] = (__bf16)v.y;
                    Bt[(c + 2) * RSA + r] = (__bf16)v.z;
                    Bt[(c + 3) * RSA + r] = (__bf16)v.w;
                }
                __syncthreads();
                v16bf af = load_a_frag(&Hes[(wm * 16 + (lane & 15)) * RSH + k0], lane);
                v16bf bf = load_b_frag(&Bt[(wn * 16 + (lane & 15)) * RSA], lane);
                acc = __builtin_amdgcn_wmma_f32_16x16x32_bf16(false, af, false, bf,
                                                              (short)0, acc, false, false);
                __syncthreads();
            }
            int n = n0 + wn * 16 + (lane & 15);
            float b1v = b1[m * 128 + n];
            float w2v = W2[m * 128 + n];
            int rbase = wm * 16 + ((lane >> 4) << 3);
#pragma unroll
            for (int i = 0; i < 8; i++) {
                float t = acc[i] + b1v;
                t = t > 0.0f ? t : 0.0f;
                atomicAdd(&gpart[rbase + i], t * w2v);
            }
            __syncthreads();
        }
        if (tid < 64) {
            int ge = e0 + tid;
            if (ge < E_) g[(size_t)ge * 3 + m] = sigmoidf(gpart[tid] + b2[m]);
        }
        __syncthreads();
    }
}

// ---------------- per-edge dot score + degree ----------------
__global__ __launch_bounds__(256) void edge_dot_deg_kernel(const float* __restrict__ h,
                                                           const int* __restrict__ src,
                                                           const int* __restrict__ dst,
                                                           float* __restrict__ a_out,
                                                           float* __restrict__ deg, int E_) {
    int e = (blockIdx.x * blockDim.x + threadIdx.x) >> 5;
    int lane = threadIdx.x & 31;
    if (e >= E_) return;
    int s = src[e], d = dst[e];
    const float4* ps = (const float4*)(h + (size_t)s * 128);
    const float4* pd = (const float4*)(h + (size_t)d * 128);
    float4 x = ps[lane], y = pd[lane];
    float p = x.x * y.x + x.y * y.y + x.z * y.z + x.w * y.w;
#pragma unroll
    for (int o = 16; o; o >>= 1) p += __shfl_down(p, o, 32);
    if (lane == 0) { a_out[e] = p; atomicAdd(&deg[d], 1.0f); }
}

// ---------------- segment softmax passes ----------------
__global__ void segmax_kernel(const float* __restrict__ a, const float* __restrict__ g,
                              const int* __restrict__ dst, float* maxA, float* maxG, int E_) {
    int e = blockIdx.x * blockDim.x + threadIdx.x;
    if (e >= E_) return;
    int d = dst[e];
    atomicMaxF(&maxA[d], a[e]);
#pragma unroll
    for (int m = 0; m < 3; m++) atomicMaxF(&maxG[d * 3 + m], g[(size_t)e * 3 + m]);
}

__global__ void expsum_kernel(const float* __restrict__ a, const float* __restrict__ g,
                              const int* __restrict__ dst,
                              const float* __restrict__ maxA, const float* __restrict__ maxG,
                              float* alpha, float* beta, float* sumA, float* sumG, int E_) {
    int e = blockIdx.x * blockDim.x + threadIdx.x;
    if (e >= E_) return;
    int d = dst[e];
    float ea = expf(a[e] - maxA[d]);
    alpha[e] = ea;
    atomicAdd(&sumA[d], ea);
#pragma unroll
    for (int m = 0; m < 3; m++) {
        float eg = expf(g[(size_t)e * 3 + m] - maxG[d * 3 + m]);
        beta[(size_t)e * 3 + m] = eg;
        atomicAdd(&sumG[d * 3 + m], eg);
    }
}

// ---------------- message kernel: gathered WMMA GEMMs + weighted scatter ----------------
// hcat[d, 0:128]   += alpha_e * (h[src] @ Wsum)          (1/K folded into Wsum)
// hcat[d, 128:256] += sum_m (beta_em/M) * (h[src] @ Wrel_m)
__global__ __launch_bounds__(256) void message_kernel(const float* __restrict__ h,
                                                      const int* __restrict__ src,
                                                      const int* __restrict__ dst,
                                                      const float* __restrict__ Wsum,  // [128,128]
                                                      const float* __restrict__ Wrel,  // [3,128,128]
                                                      const float* __restrict__ alpha_raw,
                                                      const float* __restrict__ beta_raw,
                                                      const float* __restrict__ sumA,
                                                      const float* __restrict__ sumG,
                                                      float* __restrict__ hcat, int E_) {
    __shared__ __align__(16) __bf16 Hs[64 * RSB];
    __shared__ __align__(16) __bf16 Bt4[4][32 * RSA];
    __shared__ int   srcs[64];
    __shared__ int   dsts[64];
    __shared__ float al[64];
    __shared__ float bt[64][3];
    int tid = threadIdx.x, lane = tid & 31, wid = tid >> 5;
    int wm = wid >> 1, wn = wid & 1;
    int e0 = blockIdx.x * 64;

    if (tid < 64) {
        int ge = e0 + tid;
        if (ge < E_) {
            int d = dst[ge];
            srcs[tid] = src[ge];
            dsts[tid] = d;
            al[tid] = alpha_raw[ge] / sumA[d];
#pragma unroll
            for (int m = 0; m < 3; m++)
                bt[tid][m] = beta_raw[(size_t)ge * 3 + m] / (sumG[d * 3 + m] * 3.0f);
        } else {
            srcs[tid] = -1; dsts[tid] = -1; al[tid] = 0.0f;
            bt[tid][0] = bt[tid][1] = bt[tid][2] = 0.0f;
        }
    }
    __syncthreads();
    // gather hs tile 64x128: float4 gathers + b64 LDS stores
#pragma unroll
    for (int i = 0; i < 8; i++) {
        int f = tid + i * 256;                    // 2048 float4
        int r = f >> 5, c = (f & 31) << 2;
        int s = srcs[r];
        float4 v = make_float4(0.f, 0.f, 0.f, 0.f);
        if (s >= 0) v = *(const float4*)(h + (size_t)s * 128 + c);
        store4bf(&Hs[r * RSB + c], v);
    }
    __syncthreads();

    for (int nt = 0; nt < 4; nt++) {
        int n0 = nt * 32;
        v8f acc[4];
#pragma unroll
        for (int m = 0; m < 4; m++)
#pragma unroll
            for (int i = 0; i < 8; i++) acc[m][i] = 0.0f;
        for (int kt = 0; kt < 4; kt++) {
            int k0 = kt * 32;
            // 4 weight tiles (Wsum + 3 rel), transposed into LDS
#pragma unroll
            for (int i = 0; i < 4; i++) {
                int f = tid + i * 256;            // 1024 float4
                int mm = f >> 8, rem = f & 255;
                int r = rem >> 3, c = (rem & 7) << 2;  // r=k, c=n
                const float* Wp = (mm == 0) ? Wsum : (Wrel + (size_t)(mm - 1) * 16384);
                float4 v = *(const float4*)(Wp + (size_t)(k0 + r) * 128 + n0 + c);
                Bt4[mm][(c + 0) * RSA + r] = (__bf16)v.x;
                Bt4[mm][(c + 1) * RSA + r] = (__bf16)v.y;
                Bt4[mm][(c + 2) * RSA + r] = (__bf16)v.z;
                Bt4[mm][(c + 3) * RSA + r] = (__bf16)v.w;
            }
            __syncthreads();
            v16bf af = load_a_frag(&Hs[(wm * 16 + (lane & 15)) * RSB + k0], lane);
#pragma unroll
            for (int m = 0; m < 4; m++) {
                v16bf bf = load_b_frag(&Bt4[m][(wn * 16 + (lane & 15)) * RSA], lane);
                acc[m] = __builtin_amdgcn_wmma_f32_16x16x32_bf16(false, af, false, bf,
                                                                 (short)0, acc[m], false, false);
            }
            __syncthreads();
        }
        int n = n0 + wn * 16 + (lane & 15);
        int rbase = wm * 16 + ((lane >> 4) << 3);
#pragma unroll
        for (int i = 0; i < 8; i++) {
            int r = rbase + i;
            int d = dsts[r];
            if (d >= 0) {
                float va = acc[0][i] * al[r];
                float vr = acc[1][i] * bt[r][0] + acc[2][i] * bt[r][1] + acc[3][i] * bt[r][2];
                atomicAdd(&hcat[(size_t)d * 256 + n], va);
                atomicAdd(&hcat[(size_t)d * 256 + 128 + n], vr);
            }
        }
    }
}

// ---------------- finalize: gate blend ----------------
__global__ __launch_bounds__(256) void finalize_kernel(const float* __restrict__ x,
                                                       const float* __restrict__ h,
                                                       const float* __restrict__ hred,
                                                       const float* __restrict__ deg,
                                                       const float* __restrict__ gate_W,
                                                       const float* __restrict__ gate_b,
                                                       float* __restrict__ out, int N_) {
    int n = (blockIdx.x * blockDim.x + threadIdx.x) >> 5;
    int lane = threadIdx.x & 31;
    if (n >= N_) return;
    const float4* xp = (const float4*)(x + (size_t)n * 128);
    const float4* wp = (const float4*)gate_W;
    float4 x4 = xp[lane], w4 = wp[lane];
    float p = x4.x * w4.x + x4.y * w4.y + x4.z * w4.z + x4.w * w4.w;
#pragma unroll
    for (int o = 16; o; o >>= 1) p += __shfl_xor(p, o, 32);
    float gv = sigmoidf(p + gate_b[0]);
    bool use_red = deg[n] > 0.0f;
    const float4* hp = (const float4*)((use_red ? hred : h) + (size_t)n * 128);
    float4 h4 = hp[lane];
    float4 o4;
    o4.x = gv * h4.x + (1.0f - gv) * x4.x;
    o4.y = gv * h4.y + (1.0f - gv) * x4.y;
    o4.z = gv * h4.z + (1.0f - gv) * x4.z;
    o4.w = gv * h4.w + (1.0f - gv) * x4.w;
    ((float4*)(out + (size_t)n * 128))[lane] = o4;
}

// ---------------- launch ----------------
extern "C" void kernel_launch(void* const* d_in, const int* in_sizes, int n_in,
                              void* d_out, int out_size, void* d_ws, size_t ws_size,
                              hipStream_t stream) {
    const float* nodeF      = (const float*)d_in[0];
    const float* edgeF      = (const float*)d_in[1];
    const int*   src        = (const int*)d_in[2];
    const int*   dst        = (const int*)d_in[3];
    const float* node_att_W = (const float*)d_in[4];
    const float* rel_att_W  = (const float*)d_in[5];
    const float* rel_att_W1 = (const float*)d_in[6];
    const float* rel_att_b1 = (const float*)d_in[7];
    const float* rel_att_W2 = (const float*)d_in[8];
    const float* rel_att_b2 = (const float*)d_in[9];
    const float* node_fc_W  = (const float*)d_in[10];
    const float* node_fc_b  = (const float*)d_in[11];
    const float* edge_fc_W  = (const float*)d_in[12];
    const float* edge_fc_b  = (const float*)d_in[13];
    const float* out_fc_W   = (const float*)d_in[14];
    const float* out_fc_b   = (const float*)d_in[15];
    const float* gate_W     = (const float*)d_in[16];
    const float* gate_b     = (const float*)d_in[17];

    int N_ = in_sizes[0] / 128;
    int E_ = in_sizes[2];

    float* ws = (float*)d_ws;
    size_t off = 0;
    float* h     = ws + off; off += (size_t)N_ * 128;
    float* he    = ws + off; off += (size_t)E_ * 64;
    float* gg    = ws + off; off += (size_t)E_ * 3;
    float* aa    = ws + off; off += (size_t)E_;
    float* alpha = ws + off; off += (size_t)E_;
    float* beta  = ws + off; off += (size_t)E_ * 3;
    float* maxA  = ws + off; off += (size_t)N_;
    float* sumA  = ws + off; off += (size_t)N_;
    float* maxG  = ws + off; off += (size_t)N_ * 3;
    float* sumG  = ws + off; off += (size_t)N_ * 3;
    float* deg   = ws + off; off += (size_t)N_;
    float* hcat  = ws + off; off += (size_t)N_ * 256;
    float* hred  = ws + off; off += (size_t)N_ * 128;
    float* Wsum  = ws + off; off += (size_t)128 * 128;
    float* out   = (float*)d_out;

    long tot = (long)N_ * 256;
    init_kernel<<<(unsigned)((tot + 255) / 256), 256, 0, stream>>>(hcat, deg, maxA, sumA, maxG, sumG, N_);
    wsum_kernel<<<64, 256, 0, stream>>>(node_att_W, Wsum);

    // h = node_features @ node_fc_W + b     [N,128]x[128,128]
    gemm_bias_act<<<dim3((N_ + 63) / 64, 4), 256, 0, stream>>>(nodeF, node_fc_W, node_fc_b, h, N_, 128, 128, 0);
    // he = edge_features @ edge_fc_W + b    [E,64]x[64,64]
    gemm_bias_act<<<dim3((E_ + 63) / 64, 2), 256, 0, stream>>>(edgeF, edge_fc_W, edge_fc_b, he, E_, 64, 64, 0);
    // g gates (fused WMMA MLP)
    edge_gate_kernel<<<(E_ + 63) / 64, 256, 0, stream>>>(he, rel_att_W1, rel_att_b1, rel_att_W2, rel_att_b2, gg, E_);
    // per-edge dot + degree
    edge_dot_deg_kernel<<<(unsigned)(((size_t)E_ * 32 + 255) / 256), 256, 0, stream>>>(h, src, dst, aa, deg, E_);
    // segment softmax (a and g)
    segmax_kernel<<<(E_ + 255) / 256, 256, 0, stream>>>(aa, gg, dst, maxA, maxG, E_);
    expsum_kernel<<<(E_ + 255) / 256, 256, 0, stream>>>(aa, gg, dst, maxA, maxG, alpha, beta, sumA, sumG, E_);
    // gathered WMMA message GEMMs + atomic scatter
    message_kernel<<<(E_ + 63) / 64, 256, 0, stream>>>(h, src, dst, Wsum, rel_att_W, alpha, beta, sumA, sumG, hcat, E_);
    // h_red = relu(hcat @ out_fc_W + b)     [N,256]x[256,128]
    gemm_bias_act<<<dim3((N_ + 63) / 64, 4), 256, 0, stream>>>(hcat, out_fc_W, out_fc_b, hred, N_, 256, 128, 1);
    // gated blend
    finalize_kernel<<<(unsigned)(((size_t)N_ * 32 + 255) / 256), 256, 0, stream>>>(nodeF, h, hred, deg, gate_W, gate_b, out, N_);
}
